// EdgeGraphConvolutionalLayer_81243601371200
// MI455X (gfx1250) — compile-verified, matrix operand order checked by reading, not run on hardware
//
#include <hip/hip_runtime.h>

typedef __attribute__((ext_vector_type(16))) __bf16 v16bf;
typedef __attribute__((ext_vector_type(8)))  float  v8f;
typedef __attribute__((ext_vector_type(4)))  float  v4f;

#define ALPHA_  0.02f
#define LN_EPS_ 1e-5f
#define B_ 2
#define E_ 8
#define N_ 2048
#define F_ 32

// ---------------------------------------------------------------------------
// Pass 1: r[b,e,n] = rsqrt(1 + sum_m adj[b,e,n,m])   (the +1 is the self loop)
// One wave32 per row; float4 streaming loads; shfl-xor reduction.
// ---------------------------------------------------------------------------
__global__ void __launch_bounds__(256) rowsum_rsqrt_kernel(
    const float* __restrict__ adj, float* __restrict__ r) {
  int wave = threadIdx.x >> 5;
  int lane = threadIdx.x & 31;
  long row = (long)blockIdx.x * 8 + wave;          // 0 .. B*E*N-1
  const float* p = adj + row * N_;
  float s = 0.f;
#pragma unroll 4
  for (int c = lane * 4; c < N_; c += 32 * 4) {
    v4f v = *(const v4f*)(p + c);
    s += v[0] + v[1] + v[2] + v[3];
  }
#pragma unroll
  for (int off = 16; off; off >>= 1) s += __shfl_xor(s, off, 32);
  if (lane == 0) r[row] = rsqrtf(s + 1.0f);
}

// ---------------------------------------------------------------------------
// Pass 2: pack B-operand: Wh'[b,e,k,f] = bf16(r[b,e,k] * Wh[b,k,f]) stored in
// the exact per-lane v16bf layout the WMMA B operand wants, so the hot loop
// does two aligned 32B vector loads per lane.
// Layout: flat = ((((be*64 + kb)*2 + t)*32 + lane)*16 + idx)
//   k = kb*32 + (idx<8 ? hb*8+idx : 16+hb*8+(idx-8)),  hb = lane>>4
//   f = t*16 + (lane&15)
// ---------------------------------------------------------------------------
__global__ void __launch_bounds__(256) pack_b_kernel(
    const float* __restrict__ Wh, const float* __restrict__ r,
    __bf16* __restrict__ bpack) {
  int flat = blockIdx.x * 256 + threadIdx.x;       // 0 .. 2^20-1
  int idx  = flat & 15;
  int lane = (flat >> 4) & 31;
  int t    = (flat >> 9) & 1;
  int kb   = (flat >> 10) & 63;
  int be   = flat >> 16;
  int b    = be >> 3;
  int hb   = lane >> 4;
  int k = kb * 32 + ((idx < 8) ? (hb * 8 + idx) : (16 + hb * 8 + (idx - 8)));
  int f = t * 16 + (lane & 15);
  float val = r[be * N_ + k] * Wh[((long)b * N_ + k) * F_ + f];
  bpack[flat] = (__bf16)val;
}

// ---------------------------------------------------------------------------
// Pass 3: h[b,e,n,0:32] = r[n] * ( sum_m adj[n,m]*Wh'[m,f] + r[n]*Wh[n,f] )
// 256 blocks (16 be * 16 m-blocks) x 8 waves; each wave owns a 16x32 tile.
// K-loop: 16 bodies of 4x32 K-steps; all 24 loads (512 B per lane) of a body are issued
// before any WMMA consumes them (deep MLP: ~16KB in flight per wave, so 2048
// waves cover HBM latency at 23.3 TB/s).
// ---------------------------------------------------------------------------
__global__ void __launch_bounds__(256) spmm_wmma_kernel(
    const float* __restrict__ adj, const __bf16* __restrict__ bpack,
    const float* __restrict__ r, const float* __restrict__ Wh,
    float* __restrict__ h) {
  int be   = blockIdx.x >> 4;                      // 0..15
  int mblk = blockIdx.x & 15;                      // 128 rows each
  int wave = threadIdx.x >> 5;
  int lane = threadIdx.x & 31;
  int hb   = lane >> 4;
  int rowBase = mblk * 128 + wave * 16;
  int row     = rowBase + (lane & 15);

  const float*  arow = adj + ((long)be * N_ + row) * N_;
  const __bf16* bp   = bpack + (long)be * (64 * 2 * 32 * 16);

  v8f clo = {};
  v8f chi = {};

  for (int kb0 = 0; kb0 < 64; kb0 += 4) {
    // ---- issue ALL loads for 4 K-steps first (16 x b128 + 8 x 32B per lane) ----
    v4f   a0[4], a1[4], a2[4], a3[4];
    v16bf blo[4], bhi[4];
#pragma unroll
    for (int u = 0; u < 4; ++u) {
      const float* s0 = arow + (kb0 + u) * 32 + hb * 8;  // K = k0+hb*8   ..+7
      const float* s1 = s0 + 16;                         // K = k0+hb*8+16..+7
      a0[u] = *(const v4f*)(s0);
      a1[u] = *(const v4f*)(s0 + 4);
      a2[u] = *(const v4f*)(s1);
      a3[u] = *(const v4f*)(s1 + 4);
      blo[u] = *(const v16bf*)(bp + (((long)(kb0 + u) * 2 + 0) * 32 + lane) * 16);
      bhi[u] = *(const v16bf*)(bp + (((long)(kb0 + u) * 2 + 1) * 32 + lane) * 16);
    }
    // ---- convert + 8 WMMAs ----
#pragma unroll
    for (int u = 0; u < 4; ++u) {
      v16bf a;
#pragma unroll
      for (int i = 0; i < 4; ++i) {
        a[i]      = (__bf16)a0[u][i];
        a[4 + i]  = (__bf16)a1[u][i];
        a[8 + i]  = (__bf16)a2[u][i];
        a[12 + i] = (__bf16)a3[u][i];
      }
      clo = __builtin_amdgcn_wmma_f32_16x16x32_bf16(false, a, false, blo[u],
                                                    (short)0, clo, false, false);
      chi = __builtin_amdgcn_wmma_f32_16x16x32_bf16(false, a, false, bhi[u],
                                                    (short)0, chi, false, false);
    }
  }

  // Epilogue: C/D layout: VGPR i -> M = i + 8*hb, N = lane&15
  int b  = be >> 3;
  int f0 = lane & 15;
#pragma unroll
  for (int i = 0; i < 8; ++i) {
    int rr = rowBase + i + hb * 8;
    float rs = r[be * N_ + rr];
    const float* whrow = Wh + ((long)b * N_ + rr) * F_;
    float*       hrow  = h + ((long)be * N_ + rr) * F_;
    hrow[f0]      = rs * (clo[i] + rs * whrow[f0]);
    hrow[f0 + 16] = rs * (chi[i] + rs * whrow[f0 + 16]);
  }
}

// ---------------------------------------------------------------------------
// Pass 4: the raw reshape [B,E,N,F]->[B,N,E*F] makes reshaped row n2 equal the
// contiguous block h[b, e=n2>>8, 8q:8q+8, :] (q=n2&255).  One wave per output
// row: lane f does the 256->1 dot with W_weight[f,:], then LayerNorm across
// the 32 lanes (wave32 == F) and LeakyReLU.
// ---------------------------------------------------------------------------
__global__ void __launch_bounds__(256) linear_ln_kernel(
    const float* __restrict__ h, const float* __restrict__ Ww,
    const float* __restrict__ Wb, float* __restrict__ out) {
  int gw   = blockIdx.x * 8 + (threadIdx.x >> 5);  // global wave id = row id
  int lane = threadIdx.x & 31;                     // = output feature f
  int b  = gw >> 11;
  int n2 = gw & 2047;
  int e  = n2 >> 8;
  int q  = n2 & 255;
  const float* chunk = h + (((long)(b * E_ + e)) * N_ + 8 * q) * F_;
  const float* wrow  = Ww + lane * 256;

  float acc = Wb[lane];
#pragma unroll 4
  for (int c = 0; c < 256; c += 4) {
    v4f hv = *(const v4f*)(chunk + c);
    v4f wv = *(const v4f*)(wrow + c);
    acc += hv[0] * wv[0] + hv[1] * wv[1] + hv[2] * wv[2] + hv[3] * wv[3];
  }

  float mu = acc;
#pragma unroll
  for (int off = 16; off; off >>= 1) mu += __shfl_xor(mu, off, 32);
  mu *= (1.0f / 32.0f);
  float d  = acc - mu;
  float vv = d * d;
#pragma unroll
  for (int off = 16; off; off >>= 1) vv += __shfl_xor(vv, off, 32);
  vv *= (1.0f / 32.0f);
  float y = d * rsqrtf(vv + LN_EPS_);
  y = (y >= 0.f) ? y : ALPHA_ * y;
  out[((long)b * N_ + n2) * F_ + lane] = y;
}

// ---------------------------------------------------------------------------
extern "C" void kernel_launch(void* const* d_in, const int* in_sizes, int n_in,
                              void* d_out, int out_size, void* d_ws,
                              size_t ws_size, hipStream_t stream) {
  const float* adj = (const float*)d_in[0];   // [B,E,N,N]
  const float* Wh  = (const float*)d_in[1];   // [B,N,F]
  const float* Ww  = (const float*)d_in[2];   // [F, E*F]
  const float* Wb  = (const float*)d_in[3];   // [F]
  float* out = (float*)d_out;                 // [B,N,F]

  char* ws = (char*)d_ws;
  float*  r     = (float*)ws;                              // 32768 f = 128 KB
  __bf16* bpack = (__bf16*)(ws + (131072));                // 2 MB
  float*  h     = (float*)(ws + (131072 + 2097152));       // 4 MB

  // Pass 1: degree rsqrt, 32768 rows, 8 waves/block
  rowsum_rsqrt_kernel<<<4096, 256, 0, stream>>>(adj, r);
  // Pass 2: pack bf16 B operand (1M elements)
  pack_b_kernel<<<4096, 256, 0, stream>>>(Wh, r, bpack);
  // Pass 3: WMMA main matmul (16 be-slices * 16 row-blocks)
  spmm_wmma_kernel<<<256, 256, 0, stream>>>(adj, bpack, r, Wh, h);
  // Pass 4: fused linear + layernorm + leakyrelu (4096 rows, 8 waves/block)
  linear_ln_kernel<<<512, 256, 0, stream>>>(h, Ww, Wb, out);
}
